// Model_6640019440518
// MI455X (gfx1250) — matrix operand order repaired and checked by reading, not compiled, not run on hardware
//
#include <hip/hip_runtime.h>
#include <hip/hip_bf16.h>

// ---------------------------------------------------------------------------
// HGNN forward for MI455X (gfx1250, wave32).
// Heavy GEMMs run on the matrix unit via v_wmma_f32_16x16x32_bf16 (bf16 A/B,
// f32 accumulate), 128x128 block tile, 8 WMMAs per wave per K-step, with
// double-buffered LDS tiles so global loads overlap the WMMA stream.
// Segment-mean aggregations are bandwidth-bound scatter/atomic kernels.
// ---------------------------------------------------------------------------

typedef __attribute__((ext_vector_type(16))) __bf16 v16bf;
typedef __attribute__((ext_vector_type(8)))  float  v8f;

#define N_NODES 50000
#define N_EDGES 100000
#define N_PAIRS 1600000
#define IN_DIM  2816
#define MAP_HID 512
#define IN_CH   512
#define HID_CH  256
#define OUT_CH  128

union FragU { uint4 q[2]; v16bf v; };

// ---------------------------------------------------------------------------
// Tiled GEMM: C[M,N] = relu?(A[M,K] @ W[K,N] + bias[N])
// Block: 256 threads (8 wave32 waves). Block tile 128x128, K-step 32.
// Waves arranged 4x2: wave (wm,wn) owns rows 32*wm..+32, cols 64*wn..+64
//   -> 2 A fragments x 4 B fragments = 8 WMMA accumulators per wave.
// LDS tiles are double-buffered: stage k+1 is fetched to registers while
// stage k feeds the WMMAs; one workgroup barrier per K-step.
// Requires: N % 128 == 0, K % 32 == 0 (true for all layers here).
// ---------------------------------------------------------------------------
__global__ __launch_bounds__(256) void gemm_bias_wmma(
    const float* __restrict__ A, const float* __restrict__ W,
    const float* __restrict__ bias, float* __restrict__ C,
    int M, int N, int K, int do_relu)
{
    __shared__ __bf16 As[2][128 * 32];   // [m][k], row-major
    __shared__ __bf16 Bs[2][128 * 32];   // [n][k], transposed (B column-major)

    const int tid  = threadIdx.x;
    const int lane = tid & 31;
    const int wave = tid >> 5;
    const int wm   = wave & 3;        // row group (0..3) -> 32 rows
    const int wn   = wave >> 2;       // col group (0..1) -> 64 cols
    const int hgrp = lane >> 4;       // lane half (ISA fragment layout)
    const int l16  = lane & 15;

    const int m0 = blockIdx.x * 128;
    const int n0 = blockIdx.y * 128;

    v8f acc[2][4];
#pragma unroll
    for (int am = 0; am < 2; ++am)
#pragma unroll
        for (int bn = 0; bn < 4; ++bn) acc[am][bn] = (v8f){};

    const int nk = K >> 5;            // number of K-steps of 32
    float4 aReg[4], bReg[4];

    // ---- prologue: fetch K-step 0 and populate stage 0 ----
#pragma unroll
    for (int i = 0; i < 4; ++i) {
        int idx = tid + 256 * i;                  // A: 1024 float4 slots
        int r = idx >> 3, c4 = (idx & 7) * 4;
        int rg = m0 + r;
        aReg[i] = make_float4(0.f, 0.f, 0.f, 0.f);
        if (rg < M) aReg[i] = *reinterpret_cast<const float4*>(A + (long long)rg * K + c4);
        int kr = idx >> 5, c4b = (idx & 31) * 4;  // B: 32x128 tile
        bReg[i] = *reinterpret_cast<const float4*>(W + (long long)kr * N + n0 + c4b);
    }
#pragma unroll
    for (int i = 0; i < 4; ++i) {
        int idx = tid + 256 * i;
        int r = idx >> 3, c4 = (idx & 7) * 4;
        union { __bf16 h[4]; uint2 u; } p;
        p.h[0] = (__bf16)aReg[i].x; p.h[1] = (__bf16)aReg[i].y;
        p.h[2] = (__bf16)aReg[i].z; p.h[3] = (__bf16)aReg[i].w;
        *reinterpret_cast<uint2*>(&As[0][r * 32 + c4]) = p.u;
        int kr = idx >> 5, c4b = (idx & 31) * 4;
        Bs[0][(c4b + 0) * 32 + kr] = (__bf16)bReg[i].x;
        Bs[0][(c4b + 1) * 32 + kr] = (__bf16)bReg[i].y;
        Bs[0][(c4b + 2) * 32 + kr] = (__bf16)bReg[i].z;
        Bs[0][(c4b + 3) * 32 + kr] = (__bf16)bReg[i].w;
    }
    __syncthreads();

    for (int ks = 0; ks < nk; ++ks) {
        const int cur = ks & 1;
        const int nxt = cur ^ 1;
        const bool more = (ks + 1) < nk;

        // ---- prefetch next K-step into registers (overlaps WMMAs below) ----
        if (more) {
            const int kk = (ks + 1) << 5;
#pragma unroll
            for (int i = 0; i < 4; ++i) {
                int idx = tid + 256 * i;
                int r = idx >> 3, c4 = (idx & 7) * 4;
                int rg = m0 + r;
                aReg[i] = make_float4(0.f, 0.f, 0.f, 0.f);
                if (rg < M)
                    aReg[i] = *reinterpret_cast<const float4*>(A + (long long)rg * K + kk + c4);
                int kr = idx >> 5, c4b = (idx & 31) * 4;
                bReg[i] = *reinterpret_cast<const float4*>(W + (long long)(kk + kr) * N + n0 + c4b);
            }
        }

        // ---- consume stage `cur`: 2 A frags x 4 B frags = 8 WMMAs ----
        FragU a[2];
#pragma unroll
        for (int am = 0; am < 2; ++am) {
            const int arow = wm * 32 + am * 16 + l16;
            a[am].q[0] = *reinterpret_cast<const uint4*>(&As[cur][arow * 32 + hgrp * 8]);
            a[am].q[1] = *reinterpret_cast<const uint4*>(&As[cur][arow * 32 + 16 + hgrp * 8]);
        }
#pragma unroll
        for (int bn = 0; bn < 4; ++bn) {
            FragU b;
            const int bcol = wn * 64 + bn * 16 + l16;
            b.q[0] = *reinterpret_cast<const uint4*>(&Bs[cur][bcol * 32 + hgrp * 8]);
            b.q[1] = *reinterpret_cast<const uint4*>(&Bs[cur][bcol * 32 + 16 + hgrp * 8]);
#pragma unroll
            for (int am = 0; am < 2; ++am)
                acc[am][bn] = __builtin_amdgcn_wmma_f32_16x16x32_bf16(
                    false, a[am].v, false, b.v, (short)0, acc[am][bn], false, false);
        }

        // ---- commit prefetched tile to stage `nxt` ----
        if (more) {
#pragma unroll
            for (int i = 0; i < 4; ++i) {
                int idx = tid + 256 * i;
                int r = idx >> 3, c4 = (idx & 7) * 4;
                union { __bf16 h[4]; uint2 u; } p;
                p.h[0] = (__bf16)aReg[i].x; p.h[1] = (__bf16)aReg[i].y;
                p.h[2] = (__bf16)aReg[i].z; p.h[3] = (__bf16)aReg[i].w;
                *reinterpret_cast<uint2*>(&As[nxt][r * 32 + c4]) = p.u;
                int kr = idx >> 5, c4b = (idx & 31) * 4;
                Bs[nxt][(c4b + 0) * 32 + kr] = (__bf16)bReg[i].x;
                Bs[nxt][(c4b + 1) * 32 + kr] = (__bf16)bReg[i].y;
                Bs[nxt][(c4b + 2) * 32 + kr] = (__bf16)bReg[i].z;
                Bs[nxt][(c4b + 3) * 32 + kr] = (__bf16)bReg[i].w;
            }
            __syncthreads();
        }
    }

    // ---- epilogue: bias (+relu), scatter per C/D fragment layout ----
#pragma unroll
    for (int bn = 0; bn < 4; ++bn) {
        const int gn = n0 + wn * 64 + bn * 16 + l16;
        const float bv = bias[gn];
#pragma unroll
        for (int am = 0; am < 2; ++am) {
#pragma unroll
            for (int v = 0; v < 8; ++v) {
                int r = m0 + wm * 32 + am * 16 + hgrp * 8 + v;  // VGPR v: m = v + 8*half
                if (r < M) {
                    float x = acc[am][bn][v] + bv;
                    if (do_relu) x = x > 0.f ? x : 0.f;
                    C[(long long)r * N + gn] = x;
                }
            }
        }
    }
}

// ---------------------------------------------------------------------------
// Aggregation helpers (memory-bound)
// ---------------------------------------------------------------------------
__global__ void zero_f32(float* __restrict__ p, long long n)
{
    long long i = (long long)blockIdx.x * blockDim.x + threadIdx.x;
    long long stride = (long long)gridDim.x * blockDim.x;
    for (; i < n; i += stride) p[i] = 0.f;
}

__global__ void count_pairs(const int* __restrict__ idx, int n, float* __restrict__ cnt)
{
    int i = blockIdx.x * blockDim.x + threadIdx.x;
    if (i < n) atomicAdd(&cnt[idx[i]], 1.0f);
}

// out[didx[p]][c..c+3] += X[sidx[p]][c..c+3]
__global__ void scatter_add4(const float* __restrict__ X,
                             const int* __restrict__ sidx,
                             const int* __restrict__ didx,
                             int npairs, int C,
                             float* __restrict__ out)
{
    const int C4 = C >> 2;
    long long t = (long long)blockIdx.x * blockDim.x + threadIdx.x;
    if (t >= (long long)npairs * C4) return;
    int pair = (int)(t / C4);
    int c    = (int)(t % C4) * 4;
    int s = sidx[pair];
    int d = didx[pair];
    float4 v = *reinterpret_cast<const float4*>(X + (long long)s * C + c);
    float* o = out + (long long)d * C + c;
    atomicAdd(o + 0, v.x);
    atomicAdd(o + 1, v.y);
    atomicAdd(o + 2, v.z);
    atomicAdd(o + 3, v.w);
}

__global__ void div_rows(float* __restrict__ buf, const float* __restrict__ cnt,
                         int R, int C, int do_relu)
{
    long long t = (long long)blockIdx.x * blockDim.x + threadIdx.x;
    if (t >= (long long)R * C) return;
    int r = (int)(t / C);
    float d = cnt[r];
    d = d < 1.f ? 1.f : d;
    float v = buf[t] / d;
    if (do_relu && v < 0.f) v = 0.f;
    buf[t] = v;
}

// ---------------------------------------------------------------------------
static inline int cdiv_i(long long a, long long b) { return (int)((a + b - 1) / b); }

extern "C" void kernel_launch(void* const* d_in, const int* in_sizes, int n_in,
                              void* d_out, int out_size, void* d_ws, size_t ws_size,
                              hipStream_t stream)
{
    (void)in_sizes; (void)n_in; (void)out_size; (void)ws_size;

    const float* m_emb = (const float*)d_in[0];
    const int*   vpos  = (const int*)d_in[1];
    const int*   epos  = (const int*)d_in[2];
    const int*   vneg  = (const int*)d_in[3];
    const int*   eneg  = (const int*)d_in[4];
    const float* W0  = (const float*)d_in[5];  const float* b0  = (const float*)d_in[6];
    const float* W1  = (const float*)d_in[7];  const float* b1  = (const float*)d_in[8];
    const float* W2  = (const float*)d_in[9];  const float* b2  = (const float*)d_in[10];
    const float* Wt1 = (const float*)d_in[11]; const float* bt1 = (const float*)d_in[12];
    const float* Wt2 = (const float*)d_in[13]; const float* bt2 = (const float*)d_in[14];

    float* out = (float*)d_out;
    float* ws  = (float*)d_ws;

    // workspace layout (floats); total ~89.9M floats (~360 MB)
    float* bufA = ws;                    // 25.6M : h0 -> feat -> Y1 sums
    float* bufB = ws + 25600000LL;       // 25.6M : h1 -> T1 (feat@Wt1+bt1)
    float* X1p  = ws + 51200000LL;       // 12.8M
    float* X1n  = ws + 64000000LL;       // 12.8M
    float* T2p  = ws + 76800000LL;       //  6.4M
    float* T2n  = ws + 83200000LL;       //  6.4M
    float* cEp  = ws + 89600000LL;       // 100K
    float* cEn  = cEp + N_EDGES;
    float* cVp  = cEn + N_EDGES;
    float* cVn  = cVp + N_NODES;

    // output regions (return order: X1, X2, Y_pos, Y_neg)
    float* X1o = out;                    // 50000*128
    float* X2o = out + 6400000LL;        // 50000*128
    float* Ypo = out + 12800000LL;       // 100000*128
    float* Yno = out + 25600000LL;       // 100000*128

    const dim3 blk(256);
    const int  gM = cdiv_i(N_NODES, 128); // 391

    // ---- MLP encoder (bf16 WMMA GEMMs) ----
    gemm_bias_wmma<<<dim3(gM, MAP_HID / 128), blk, 0, stream>>>(m_emb, W0, b0, bufA, N_NODES, MAP_HID, IN_DIM, 1);
    gemm_bias_wmma<<<dim3(gM, MAP_HID / 128), blk, 0, stream>>>(bufA, W1, b1, bufB, N_NODES, MAP_HID, MAP_HID, 1);
    gemm_bias_wmma<<<dim3(gM, IN_CH  / 128), blk, 0, stream>>>(bufB, W2, b2, bufA, N_NODES, IN_CH, MAP_HID, 0);   // feat
    // T1 = feat @ Wt1 + bt1  (shared by pos and neg branches since X1==X2==feat)
    gemm_bias_wmma<<<dim3(gM, HID_CH / 128), blk, 0, stream>>>(bufA, Wt1, bt1, bufB, N_NODES, HID_CH, IN_CH, 0);

    // ---- incidence counts (reused across both layers) ----
    zero_f32<<<cdiv_i(2 * N_EDGES + 2 * N_NODES, 256), blk, 0, stream>>>(cEp, 2LL * N_EDGES + 2LL * N_NODES);
    count_pairs<<<cdiv_i(N_PAIRS, 256), blk, 0, stream>>>(epos, N_PAIRS, cEp);
    count_pairs<<<cdiv_i(N_PAIRS, 256), blk, 0, stream>>>(eneg, N_PAIRS, cEn);
    count_pairs<<<cdiv_i(N_PAIRS, 256), blk, 0, stream>>>(vpos, N_PAIRS, cVp);
    count_pairs<<<cdiv_i(N_PAIRS, 256), blk, 0, stream>>>(vneg, N_PAIRS, cVn);

    const long long eC1 = (long long)N_EDGES * HID_CH;   // 25.6M
    const long long vC1 = (long long)N_NODES * HID_CH;   // 12.8M
    const long long p1  = (long long)N_PAIRS * (HID_CH / 4);
    const long long p2  = (long long)N_PAIRS * (OUT_CH / 4);

    // ---- HGNN layer 1, pos graph (relu on e2v result) ----
    zero_f32<<<16384, blk, 0, stream>>>(bufA, eC1);
    scatter_add4<<<cdiv_i(p1, 256), blk, 0, stream>>>(bufB, vpos, epos, N_PAIRS, HID_CH, bufA);
    div_rows<<<cdiv_i(eC1, 256), blk, 0, stream>>>(bufA, cEp, N_EDGES, HID_CH, 0);      // Y1_pos
    zero_f32<<<16384, blk, 0, stream>>>(X1p, vC1);
    scatter_add4<<<cdiv_i(p1, 256), blk, 0, stream>>>(bufA, epos, vpos, N_PAIRS, HID_CH, X1p);
    div_rows<<<cdiv_i(vC1, 256), blk, 0, stream>>>(X1p, cVp, N_NODES, HID_CH, 1);

    // ---- HGNN layer 1, neg graph ----
    zero_f32<<<16384, blk, 0, stream>>>(bufA, eC1);
    scatter_add4<<<cdiv_i(p1, 256), blk, 0, stream>>>(bufB, vneg, eneg, N_PAIRS, HID_CH, bufA);
    div_rows<<<cdiv_i(eC1, 256), blk, 0, stream>>>(bufA, cEn, N_EDGES, HID_CH, 0);      // Y1_neg
    zero_f32<<<16384, blk, 0, stream>>>(X1n, vC1);
    scatter_add4<<<cdiv_i(p1, 256), blk, 0, stream>>>(bufA, eneg, vneg, N_PAIRS, HID_CH, X1n);
    div_rows<<<cdiv_i(vC1, 256), blk, 0, stream>>>(X1n, cVn, N_NODES, HID_CH, 1);

    // ---- HGNN layer 2 dense (WMMA) ----
    gemm_bias_wmma<<<dim3(gM, OUT_CH / 128), blk, 0, stream>>>(X1p, Wt2, bt2, T2p, N_NODES, OUT_CH, HID_CH, 0);
    gemm_bias_wmma<<<dim3(gM, OUT_CH / 128), blk, 0, stream>>>(X1n, Wt2, bt2, T2n, N_NODES, OUT_CH, HID_CH, 0);

    // ---- HGNN layer 2 aggregation, sums accumulated directly in d_out ----
    zero_f32<<<16384, blk, 0, stream>>>(out, 38400000LL);

    const long long eC2 = (long long)N_EDGES * OUT_CH;   // 12.8M
    const long long vC2 = (long long)N_NODES * OUT_CH;   //  6.4M

    scatter_add4<<<cdiv_i(p2, 256), blk, 0, stream>>>(T2p, vpos, epos, N_PAIRS, OUT_CH, Ypo);
    div_rows<<<cdiv_i(eC2, 256), blk, 0, stream>>>(Ypo, cEp, N_EDGES, OUT_CH, 0);       // Y_pos
    scatter_add4<<<cdiv_i(p2, 256), blk, 0, stream>>>(Ypo, epos, vpos, N_PAIRS, OUT_CH, X1o);
    div_rows<<<cdiv_i(vC2, 256), blk, 0, stream>>>(X1o, cVp, N_NODES, OUT_CH, 0);       // X1

    scatter_add4<<<cdiv_i(p2, 256), blk, 0, stream>>>(T2n, vneg, eneg, N_PAIRS, OUT_CH, Yno);
    div_rows<<<cdiv_i(eC2, 256), blk, 0, stream>>>(Yno, cEn, N_EDGES, OUT_CH, 0);       // Y_neg
    scatter_add4<<<cdiv_i(p2, 256), blk, 0, stream>>>(Yno, eneg, vneg, N_PAIRS, OUT_CH, X2o);
    div_rows<<<cdiv_i(vC2, 256), blk, 0, stream>>>(X2o, cVn, N_NODES, OUT_CH, 0);       // X2
}
